// VanillaSLSTMFeat_36275293782360
// MI455X (gfx1250) — matrix operand order repaired and checked by reading, not compiled, not run on hardware
//
#include <hip/hip_runtime.h>
#include <hip/hip_bf16.h>

// Shaw relative-position MHA for MI455X (gfx1250), bf16 WMMA pipeline.
// T=2048 B=2 E=1024 H=16 D=64 MAX_REL=16
#define T_LEN   2048
#define B_SZ    2
#define E_DIM   1024
#define H_NUM   16
#define D_DIM   64
#define NHEAD   32            // B*H
#define MAXREL  16
#define NBUCKET 33            // 2*MAXREL+1

typedef __attribute__((ext_vector_type(16))) __bf16 v16bf;
typedef __attribute__((ext_vector_type(8)))  float  v8f;
typedef __attribute__((ext_vector_type(4)))  unsigned int u32x4;
typedef __attribute__((ext_vector_type(8)))  int          i32x8;

union BFrag { v16bf v; unsigned short u[16]; uint4 q[2]; };

static __device__ __forceinline__ unsigned short f2bf(float f) {
    union { float f; unsigned u; } c; c.f = f;
    unsigned r = c.u + 0x7fffu + ((c.u >> 16) & 1u);   // RNE
    return (unsigned short)(r >> 16);
}
static __device__ __forceinline__ float bf2f(unsigned short h) {
    union { unsigned u; float f; } c; c.u = ((unsigned)h) << 16;
    return c.f;
}
static __device__ __forceinline__ v8f vzero8() {
    v8f z;
#pragma unroll
    for (int i = 0; i < 8; ++i) z[i] = 0.0f;
    return z;
}
static __device__ __forceinline__ v8f wmma_bf16(v16bf a, v16bf b, v8f c) {
    return __builtin_amdgcn_wmma_f32_16x16x32_bf16(false, a, false, b,
                                                   (short)0, c, false, false);
}

// ---------------------------------------------------------------------------
// Kernel 0: bulk f32 -> bf16 conversion (one pass, bandwidth bound).
// ---------------------------------------------------------------------------
__global__ __launch_bounds__(256) void f32_to_bf16_kernel(
    const float* __restrict__ src, unsigned short* __restrict__ dst, int n)
{
    int i = (blockIdx.x * blockDim.x + threadIdx.x) * 4;
    if (i + 3 < n) {
        const float4 f = *(const float4*)(src + i);
        ushort4 o;
        o.x = f2bf(f.x); o.y = f2bf(f.y); o.z = f2bf(f.z); o.w = f2bf(f.w);
        *(ushort4*)(dst + i) = o;
    }
}

// ---------------------------------------------------------------------------
// Kernel 1: fused Q/K/V projections from pre-converted bf16 operands.
// out = X @ W.T + b ; q additionally scaled by D^-0.5.
// q_ws,k_ws: bf16 [NHEAD][T][D]; v_ws: bf16 [NHEAD][D][T] (transposed).
// One wave computes a 16x64 tile; 8 waves / block.
// ---------------------------------------------------------------------------
__global__ __launch_bounds__(256) void qkv_proj_kernel(
    const unsigned short* __restrict__ Xbf,
    const unsigned short* __restrict__ Wqb, const float* __restrict__ bq,
    const unsigned short* __restrict__ Wkb, const float* __restrict__ bk,
    const unsigned short* __restrict__ Wvb, const float* __restrict__ bv,
    unsigned short* __restrict__ q_ws,
    unsigned short* __restrict__ k_ws,
    unsigned short* __restrict__ v_ws)
{
    const int mat  = blockIdx.z;                         // 0=q 1=k 2=v
    const unsigned short* W = (mat == 0) ? Wqb : ((mat == 1) ? Wkb : Wvb);
    const float* bias       = (mat == 0) ? bq  : ((mat == 1) ? bk  : bv);
    const int lane = threadIdx.x & 31;
    const int wave = threadIdx.x >> 5;
    const int i0   = blockIdx.x * 128 + wave * 16;       // output row base
    const int j0   = blockIdx.y * 64;                    // output col base
    const int ln   = lane & 15;
    const int half = lane >> 4;

    v8f acc[4];
#pragma unroll
    for (int t = 0; t < 4; ++t) acc[t] = vzero8();

    for (int e0 = 0; e0 < E_DIM; e0 += 32) {
        // A fragment: rows i0..i0+15, K = e0..e0+31
        BFrag a;
        const unsigned short* p1 = Xbf + (size_t)(i0 + ln) * E_DIM + e0 + half * 8;
        a.q[0] = *(const uint4*)p1;          // K = 8*half + 0..7
        a.q[1] = *(const uint4*)(p1 + 16);   // K = 16 + 8*half + 0..7
#pragma unroll
        for (int jt = 0; jt < 4; ++jt) {
            // B fragment: col j0+jt*16+ln, K = e0 + 8*half + 0..7 and 16 + 8*half + 0..7
            BFrag b;
            const unsigned short* pw =
                W + (size_t)(j0 + jt * 16 + ln) * E_DIM + e0 + half * 8;
            b.q[0] = *(const uint4*)pw;
            b.q[1] = *(const uint4*)(pw + 16);
            acc[jt] = wmma_bf16(a.v, b.v, acc[jt]);
        }
    }

    const float scale = (mat == 0) ? 0.125f : 1.0f;      // D^-0.5 = 1/8
#pragma unroll
    for (int jt = 0; jt < 4; ++jt) {
#pragma unroll
        for (int r = 0; r < 8; ++r) {
            const int row = half * 8 + r;
            const int i   = i0 + row;                    // i = t*B + b
            const int t   = i >> 1, bb = i & 1;
            const int j   = j0 + jt * 16 + ln;
            const int h   = j >> 6, dd = j & 63;
            const int head = bb * H_NUM + h;
            const float val = (acc[jt][r] + bias[j]) * scale;
            const unsigned short bw = f2bf(val);
            if (mat == 2)      v_ws[((size_t)head * D_DIM + dd) * T_LEN + t] = bw;
            else if (mat == 1) k_ws[((size_t)head * T_LEN + t) * D_DIM + dd] = bw;
            else               q_ws[((size_t)head * T_LEN + t) * D_DIM + dd] = bw;
        }
    }
}

// ---------------------------------------------------------------------------
// Kernel 2: flash attention with Shaw relative positions.
// One wave per (head, 16 query rows). Online softmax; relpos-k via 33-entry
// qr LUT; relpos-v via diagonal-band score cache + low/high tail sums.
// K tiles staged into LDS by the Tensor Data Mover (tensor_load_to_lds).
// attn_ws: bf16 [T][B][E].
// ---------------------------------------------------------------------------
__global__ __launch_bounds__(32) void attn_kernel(
    const unsigned short* __restrict__ q_ws,
    const unsigned short* __restrict__ k_ws,
    const unsigned short* __restrict__ v_ws,
    const float* __restrict__ rk_table,
    const float* __restrict__ rv_table,
    unsigned short* __restrict__ attn_ws)
{
    __shared__ float rks[NBUCKET * D_DIM];
    __shared__ float rvs[NBUCKET * D_DIM];
    __shared__ float qr[16][NBUCKET];
    __shared__ float band[16][32];                 // raw scores for |s-t|<=15
    __shared__ __attribute__((aligned(16))) unsigned short ptile[16][32];
    __shared__ __attribute__((aligned(16))) unsigned short kstage[32 * D_DIM]; // TDM dest

    const int lane = threadIdx.x;
    const int t0   = blockIdx.x * 16;
    const int head = blockIdx.y;
    const int ln   = lane & 15;
    const int half = lane >> 4;

    for (int idx = lane; idx < NBUCKET * D_DIM; idx += 32) {
        rks[idx] = rk_table[idx];
        rvs[idx] = rv_table[idx];
    }
    for (int idx = lane; idx < 16 * 32; idx += 32)
        band[idx >> 5][idx & 31] = -3.0e38f;
    __syncthreads();

    // qr[m][j] = q[m,:] . rk_table[j,:]
    const unsigned short* qbase = q_ws + ((size_t)head * T_LEN + t0) * D_DIM;
    for (int idx = lane; idx < 16 * NBUCKET; idx += 32) {
        const int mm = idx / NBUCKET, j = idx % NBUCKET;
        float s = 0.f;
        for (int d = 0; d < D_DIM; ++d)
            s += bf2f(qbase[mm * D_DIM + d]) * rks[j * D_DIM + d];
        qr[mm][j] = s;
    }
    __syncthreads();

    // Q A-fragments (16x32 each, K = d 0..31 and 32..63)
    BFrag aq0, aq1;
    {
        const unsigned short* p = qbase + ln * D_DIM;
        aq0.q[0] = *(const uint4*)(p + half * 8);
        aq0.q[1] = *(const uint4*)(p + 16 + half * 8);
        aq1.q[0] = *(const uint4*)(p + 32 + half * 8);
        aq1.q[1] = *(const uint4*)(p + 48 + half * 8);
    }

    v8f o[4];
#pragma unroll
    for (int t = 0; t < 4; ++t) o[t] = vzero8();
    float mrow[8], lsum[8], hsum[8];
#pragma unroll
    for (int r = 0; r < 8; ++r) { mrow[r] = -3.0e38f; lsum[r] = 0.f; hsum[r] = 0.f; }

    const unsigned short* kb = k_ws + (size_t)head * T_LEN * D_DIM;
    const unsigned short* vb = v_ws + (size_t)head * D_DIM * T_LEN;

    // Static part of the Tensor DMA descriptor (D#), group 1:
    //   data_size=2B, tensor_dim0=64, tensor_dim1=2048, tile 64x32, stride0=64
    i32x8 g1;
    g1[0] = 1 << 16;                                   // data_size code 1 (2B)
    g1[1] = (int)((64u & 0xffffu) << 16);              // tensor_dim0 lo16
    g1[2] = (int)((2048u & 0xffffu) << 16);            // dim0 hi16 | dim1 lo16
    g1[3] = (int)(64u << 16);                          // dim1 hi16 | tile_dim0=64
    g1[4] = 32;                                        // tile_dim1=32, tile_dim2=0
    g1[5] = 64;                                        // tensor_dim0_stride lo32
    g1[6] = 0;                                         // stride0 hi | stride1 lo
    g1[7] = 0;
    const unsigned lds_off = (unsigned)(uintptr_t)(&kstage[0]);

    for (int s0 = 0; s0 < T_LEN; s0 += 32) {
        // ---- TDM: stage K tile [s0..s0+31][0..63] (bf16) into LDS
        {
            const unsigned long long ga =
                (unsigned long long)(uintptr_t)(kb + (size_t)s0 * D_DIM);
            u32x4 g0;
            g0[0] = 1u;                                // count=1 (valid descriptor)
            g0[1] = lds_off;                           // lds_addr
            g0[2] = (unsigned)(ga & 0xffffffffu);      // global_addr lo
            g0[3] = (unsigned)((ga >> 32) & 0x01ffffffu) | (2u << 30); // hi | type=2
            asm volatile("tensor_load_to_lds %0, %1" :: "s"(g0), "s"(g1) : "memory");
        }
        // prefetch next V tile while TDM runs
        if (s0 + 32 < T_LEN)
            __builtin_prefetch(vb + (size_t)ln * T_LEN + s0 + 32, 0, 1);
        __builtin_amdgcn_s_wait_tensorcnt(0);

        // ---- scores: two 16-col tiles, contract over d=64 (2 WMMAs each)
        v8f sc[2];
        sc[0] = vzero8(); sc[1] = vzero8();
#pragma unroll
        for (int c = 0; c < 2; ++c) {
            BFrag bk0, bk1;
            const unsigned short* p = &kstage[(c * 16 + ln) * D_DIM + half * 8];
            bk0.q[0] = *(const uint4*)p;
            bk0.q[1] = *(const uint4*)(p + 16);
            bk1.q[0] = *(const uint4*)(p + 32);
            bk1.q[1] = *(const uint4*)(p + 48);
            sc[c] = wmma_bf16(aq0.v, bk0.v, sc[c]);
            sc[c] = wmma_bf16(aq1.v, bk1.v, sc[c]);
        }

        // ---- rel-k add, band cache, online-softmax stats, P tile
        float fac[8];
#pragma unroll
        for (int r = 0; r < 8; ++r) {
            const int row  = half * 8 + r;
            const int trow = t0 + row;
            float v0 = sc[0][r], v1 = sc[1][r];
            const int d0 = (s0 + ln) - trow;
            const int d1 = d0 + 16;
            const int j0i = d0 < -MAXREL ? 0 : (d0 > MAXREL ? 2 * MAXREL : d0 + MAXREL);
            const int j1i = d1 < -MAXREL ? 0 : (d1 > MAXREL ? 2 * MAXREL : d1 + MAXREL);
            v0 += qr[row][j0i];
            v1 += qr[row][j1i];
            if (d0 >= -15 && d0 <= 15) band[row][d0 + 15] = v0;
            if (d1 >= -15 && d1 <= 15) band[row][d1 + 15] = v1;

            float vmax = fmaxf(v0, v1);
            vmax = fmaxf(vmax, __shfl_xor(vmax, 1, 32));
            vmax = fmaxf(vmax, __shfl_xor(vmax, 2, 32));
            vmax = fmaxf(vmax, __shfl_xor(vmax, 4, 32));
            vmax = fmaxf(vmax, __shfl_xor(vmax, 8, 32));
            const float mnew = fmaxf(mrow[r], vmax);
            fac[r]  = __expf(mrow[r] - mnew);
            mrow[r] = mnew;

            const float p0 = __expf(v0 - mnew);
            const float p1 = __expf(v1 - mnew);
            lsum[r] = lsum[r] * fac[r] + p0 + p1;
            float hs = hsum[r] * fac[r];
            if (d0 >= MAXREL) hs += p0;
            if (d1 >= MAXREL) hs += p1;
            hsum[r] = hs;

            ptile[row][ln]      = f2bf(p0);
            ptile[row][16 + ln] = f2bf(p1);
            o[0][r] *= fac[r]; o[1][r] *= fac[r]; o[2][r] *= fac[r]; o[3][r] *= fac[r];
        }
        __syncthreads();

        // ---- P as A-fragment (16x32 over local s)
        BFrag ap;
        {
            const unsigned short* p = &ptile[ln][0];
            ap.q[0] = *(const uint4*)(p + half * 8);
            ap.q[1] = *(const uint4*)(p + 16 + half * 8);
        }
        // ---- P x V: 4 d-tiles
#pragma unroll
        for (int dt = 0; dt < 4; ++dt) {
            BFrag bvf;
            const unsigned short* p = vb + (size_t)(dt * 16 + ln) * T_LEN + s0 + half * 8;
            bvf.q[0] = *(const uint4*)p;
            bvf.q[1] = *(const uint4*)(p + 16);
            o[dt] = wmma_bf16(ap.v, bvf.v, o[dt]);
        }
        __syncthreads();   // ptile/kstage reuse
    }

    // ---- final row sums across the 16-lane half
    float ltot[8], htot[8];
#pragma unroll
    for (int r = 0; r < 8; ++r) {
        float l = lsum[r], h = hsum[r];
        l += __shfl_xor(l, 1, 32); h += __shfl_xor(h, 1, 32);
        l += __shfl_xor(l, 2, 32); h += __shfl_xor(h, 2, 32);
        l += __shfl_xor(l, 4, 32); h += __shfl_xor(h, 4, 32);
        l += __shfl_xor(l, 8, 32); h += __shfl_xor(h, 8, 32);
        ltot[r] = l; htot[r] = h;
    }
    __syncthreads();   // band visible

#pragma unroll
    for (int r = 0; r < 8; ++r) {
        const int row  = half * 8 + r;
        const float mfin = mrow[r];
        float bandtot = 0.f;
        float pb[31];
#pragma unroll
        for (int jb = 0; jb < 31; ++jb) {
            pb[jb] = __expf(band[row][jb] - mfin);
            bandtot += pb[jb];
        }
        const float low = ltot[r] - htot[r] - bandtot;
        const float inv = 1.0f / ltot[r];
#pragma unroll
        for (int dt = 0; dt < 4; ++dt) {
            const int dcol = dt * 16 + ln;
            float add = low * rvs[dcol] + htot[r] * rvs[2 * MAXREL * D_DIM + dcol];
#pragma unroll
            for (int jb = 0; jb < 31; ++jb)
                add += pb[jb] * rvs[(jb + 1) * D_DIM + dcol];
            const float val = (o[dt][r] + add) * inv;
            const int t  = t0 + row;
            const int bb = head >> 4, h = head & 15;
            attn_ws[((size_t)t * B_SZ + bb) * E_DIM + h * D_DIM + dcol] = f2bf(val);
        }
    }
}

// ---------------------------------------------------------------------------
// Kernel 3: output projection.  attn(bf16)[T*B,E] @ Wo.T + bo -> f32 out.
// ---------------------------------------------------------------------------
__global__ __launch_bounds__(256) void out_proj_kernel(
    const unsigned short* __restrict__ Xbf,
    const unsigned short* __restrict__ Wob, const float* __restrict__ bo,
    float* __restrict__ out)
{
    const int lane = threadIdx.x & 31;
    const int wave = threadIdx.x >> 5;
    const int i0   = blockIdx.x * 128 + wave * 16;
    const int j0   = blockIdx.y * 64;
    const int ln   = lane & 15;
    const int half = lane >> 4;

    v8f acc[4];
#pragma unroll
    for (int t = 0; t < 4; ++t) acc[t] = vzero8();

    for (int e0 = 0; e0 < E_DIM; e0 += 32) {
        BFrag a;
        const unsigned short* p1 = Xbf + (size_t)(i0 + ln) * E_DIM + e0 + half * 8;
        a.q[0] = *(const uint4*)p1;
        a.q[1] = *(const uint4*)(p1 + 16);
#pragma unroll
        for (int jt = 0; jt < 4; ++jt) {
            BFrag b;
            const unsigned short* pw =
                Wob + (size_t)(j0 + jt * 16 + ln) * E_DIM + e0 + half * 8;
            b.q[0] = *(const uint4*)pw;
            b.q[1] = *(const uint4*)(pw + 16);
            acc[jt] = wmma_bf16(a.v, b.v, acc[jt]);
        }
    }
#pragma unroll
    for (int jt = 0; jt < 4; ++jt) {
#pragma unroll
        for (int r = 0; r < 8; ++r) {
            const int i = i0 + half * 8 + r;
            const int j = j0 + jt * 16 + ln;
            out[(size_t)i * E_DIM + j] = acc[jt][r] + bo[j];
        }
    }
}

// ---------------------------------------------------------------------------
extern "C" void kernel_launch(void* const* d_in, const int* in_sizes, int n_in,
                              void* d_out, int out_size, void* d_ws, size_t ws_size,
                              hipStream_t stream) {
    (void)in_sizes; (void)n_in; (void)out_size; (void)ws_size;
    const float* query = (const float*)d_in[0];
    const float* Wq    = (const float*)d_in[1];
    const float* bq    = (const float*)d_in[2];
    const float* Wk    = (const float*)d_in[3];
    const float* bk    = (const float*)d_in[4];
    const float* Wv    = (const float*)d_in[5];
    const float* bv    = (const float*)d_in[6];
    const float* Wo    = (const float*)d_in[7];
    const float* bo    = (const float*)d_in[8];
    const float* rk    = (const float*)d_in[9];
    const float* rv    = (const float*)d_in[10];
    float* out = (float*)d_out;

    const size_t NTOK = (size_t)T_LEN * B_SZ;           // 4096 rows
    const size_t XSZ  = NTOK * E_DIM;                   // 4M elements
    const size_t WSZ  = (size_t)E_DIM * E_DIM;          // 1M elements
    const size_t HSZ  = (size_t)NHEAD * T_LEN * D_DIM;  // 4M elements

    unsigned short* Xbf     = (unsigned short*)d_ws;
    unsigned short* Wqb     = Xbf + XSZ;
    unsigned short* Wkb     = Wqb + WSZ;
    unsigned short* Wvb     = Wkb + WSZ;
    unsigned short* Wob     = Wvb + WSZ;
    unsigned short* q_ws    = Wob + WSZ;
    unsigned short* k_ws    = q_ws + HSZ;
    unsigned short* v_ws    = k_ws + HSZ;
    unsigned short* attn_ws = v_ws + HSZ;

    // pass 0: bulk conversions to bf16
    f32_to_bf16_kernel<<<(int)(XSZ / 4 + 255) / 256, 256, 0, stream>>>(query, Xbf, (int)XSZ);
    f32_to_bf16_kernel<<<(int)(WSZ / 4 + 255) / 256, 256, 0, stream>>>(Wq, Wqb, (int)WSZ);
    f32_to_bf16_kernel<<<(int)(WSZ / 4 + 255) / 256, 256, 0, stream>>>(Wk, Wkb, (int)WSZ);
    f32_to_bf16_kernel<<<(int)(WSZ / 4 + 255) / 256, 256, 0, stream>>>(Wv, Wvb, (int)WSZ);
    f32_to_bf16_kernel<<<(int)(WSZ / 4 + 255) / 256, 256, 0, stream>>>(Wo, Wob, (int)WSZ);

    dim3 g1((unsigned)(NTOK / 128), E_DIM / 64, 3);
    qkv_proj_kernel<<<g1, 256, 0, stream>>>(Xbf, Wqb, bq, Wkb, bk, Wvb, bv,
                                            q_ws, k_ws, v_ws);

    dim3 g2(T_LEN / 16, NHEAD);
    attn_kernel<<<g2, 32, 0, stream>>>(q_ws, k_ws, v_ws, rk, rv, attn_ws);

    dim3 g3((unsigned)(NTOK / 128), E_DIM / 64);
    out_proj_kernel<<<g3, 256, 0, stream>>>(attn_ws, Wob, bo, out);
}
